// fairCML_59373627900458
// MI455X (gfx1250) — matrix operand order repaired and checked by reading, not compiled, not run on hardware
//
#include <hip/hip_runtime.h>
#include <math.h>

typedef __attribute__((ext_vector_type(16))) __bf16 v16bf;
typedef __attribute__((ext_vector_type(8)))  float  v8f;

#define F_DIM   128
#define H_DIM   512
#define HS_PITCH 520            // padded LDS pitch (elements), 1040B = 65*16B
#define BN_EPS  1e-5f
#define MARGIN  0.5f
#define L2_REG  1e-3f

#if defined(__has_builtin)
#if __has_builtin(__builtin_amdgcn_tensor_load_to_lds) && \
    __has_builtin(__builtin_amdgcn_s_wait_tensorcnt)
#define HAVE_TDM 1
#endif
#endif

// ---------------------------------------------------------------------------
// Fragment loaders matching CDNA5 16-bit WMMA VGPR layouts (CDNA5 ISA):
// A 16x32 (MxK): lane L holds row M=L%16; lanes 0-15: K={0..7,16..23},
//                lanes 16-31: K={8..15,24..31}; 2 elems per VGPR.
// B 32x16 (KxN): lane L holds col N=L%16; lanes 0-15: K={0..7,16..23},
//                lanes 16-31: K={8..15,24..31} (same order as A).
// Both fragments read directly from row-major rows (no transpose needed):
// A from an activation row, B from a weight row (out = x @ W^T).
// ---------------------------------------------------------------------------
__device__ __forceinline__ v16bf load_frag_a(const __bf16* rowp, int hi, int k0) {
  const __bf16* p = rowp + k0 + (hi ? 8 : 0);
  union { v16bf v; uint4 u[2]; } r;
  r.u[0] = *reinterpret_cast<const uint4*>(p);        // K = s .. s+7
  r.u[1] = *reinterpret_cast<const uint4*>(p + 16);   // K = 16+s .. 16+s+7
  return r.v;
}
__device__ __forceinline__ v16bf load_frag_b(const __bf16* rowp, int hi, int k0) {
  const __bf16* p = rowp + k0 + (hi ? 8 : 0);
  union { v16bf v; uint4 u[2]; } r;
  r.u[0] = *reinterpret_cast<const uint4*>(p);        // K = t .. t+7
  r.u[1] = *reinterpret_cast<const uint4*>(p + 16);   // K = 16+t .. 16+t+7
  return r.v;
}
__device__ __forceinline__ v8f wmma_bf16(v16bf a, v16bf b, v8f c) {
  return __builtin_amdgcn_wmma_f32_16x16x32_bf16(false, a, false, b,
                                                 (short)0, c, false, false);
}

// --------------------------- kernel 1: weight convert ----------------------
__global__ __launch_bounds__(256) void wconv_kernel(const float* __restrict__ W1,
                                                    const float* __restrict__ W2,
                                                    __bf16* __restrict__ W1b,
                                                    __bf16* __restrict__ W2b) {
  int i = blockIdx.x * 256 + threadIdx.x;
  if (i < H_DIM * F_DIM) {
    W1b[i] = (__bf16)W1[i];
    W2b[i] = (__bf16)W2[i];
  }
}

// --------------------------- kernel 2: gather + renorm ---------------------
// one wave (32 lanes) per flat row; rows [0,N)=user, [N,2N)=pos, [2N,3N)=neg
__global__ __launch_bounds__(256) void gather_kernel(const int* __restrict__ user0,
                                                     const int* __restrict__ posi,
                                                     const int* __restrict__ negi,
                                                     const float* __restrict__ utab,
                                                     const float* __restrict__ itab,
                                                     __bf16* __restrict__ Xb, int N) {
  int wid  = (blockIdx.x * 256 + threadIdx.x) >> 5;
  int lane = threadIdx.x & 31;
  if (wid >= 3 * N) return;
  int n   = wid % N;
  int sel = wid / N;
  const float* src;
  if (sel == 0)      src = utab + (size_t)user0[n] * F_DIM;
  else if (sel == 1) src = itab + (size_t)posi[n] * F_DIM;
  else               src = itab + (size_t)negi[n] * F_DIM;
  float4 v = reinterpret_cast<const float4*>(src)[lane];
  float ss = v.x * v.x + v.y * v.y + v.z * v.z + v.w * v.w;
  #pragma unroll
  for (int o = 16; o > 0; o >>= 1) ss += __shfl_xor(ss, o, 32);
  float nrm = sqrtf(ss);
  float s = nrm > 1.0f ? 1.0f / nrm : 1.0f;   // max_norm=1 renorm
  __bf16* dst = Xb + (size_t)wid * F_DIM + lane * 4;
  dst[0] = (__bf16)(v.x * s); dst[1] = (__bf16)(v.y * s);
  dst[2] = (__bf16)(v.z * s); dst[3] = (__bf16)(v.w * s);
}

// --------------------------- kernel 3: fused filter (WMMA) -----------------
// one wave per 32-row tile (two 16-row M-strips sharing every B fragment):
//   TDM-stage X tile -> LDS, GEMM1(32x512x128)+bias+BN+LeakyReLU -> LDS bf16,
//   GEMM2(32x128x512)+bias -> E (f32, global)
__global__ __launch_bounds__(32) void filter_kernel(const __bf16* __restrict__ Xb,
                                                    const __bf16* __restrict__ W1b,
                                                    const __bf16* __restrict__ W2b,
                                                    const float* __restrict__ b1,
                                                    const float* __restrict__ g1,
                                                    const float* __restrict__ be1,
                                                    const float* __restrict__ rm,
                                                    const float* __restrict__ rv,
                                                    const float* __restrict__ b2,
                                                    float* __restrict__ E) {
  // single LDS blob so the TDM staging buffer is at LDS offset 0
  __shared__ __attribute__((aligned(16)))
      char smem[32 * F_DIM * 2 + 32 * HS_PITCH * 2];
  __bf16(*Xs)[F_DIM]    = reinterpret_cast<__bf16(*)[F_DIM]>(smem);
  __bf16(*Hs)[HS_PITCH] = reinterpret_cast<__bf16(*)[HS_PITCH]>(smem + 32 * F_DIM * 2);

  const int lane = threadIdx.x;
  const int lm   = lane & 15;
  const int hi   = lane >> 4;                 // 0: low K half, 1: high K half
  const int r0   = blockIdx.x * 32;           // flat row base (32 rows/tile)

  // ---- stage X tile (32 x 128 bf16) into LDS via the Tensor Data Mover ----
#if defined(HAVE_TDM)
  {
    typedef __attribute__((ext_vector_type(4))) unsigned int v4u;
    typedef __attribute__((ext_vector_type(8))) int v8i_;
    typedef __attribute__((ext_vector_type(4))) int v4i_;
    unsigned long long ga =
        (unsigned long long)(const void*)(Xb + (size_t)r0 * F_DIM);
    v4u g0;
    g0[0] = 1u;                                    // count=1, user descriptor
    g0[1] = 0u;                                    // lds_addr = 0 (Xs)
    g0[2] = (unsigned)(ga & 0xFFFFFFFFu);          // global_addr[31:0]
    g0[3] = (unsigned)((ga >> 32) & 0x01FFFFFFu)   // global_addr[56:32]
            | (2u << 30);                          // type=2 ("image")
    v8i_ g1d;
    g1d[0] = (int)(1u << 16);        // workgroup_mask=0, data_size=1 (2B)
    g1d[1] = (int)(128u << 16);      // tensor_dim0[15:0]=128 @ bits[63:48]
    g1d[2] = (int)(32u << 16);       // tensor_dim1[15:0]=32  @ bits[95:80]
    g1d[3] = (int)(128u << 16);      // tile_dim0=128         @ bits[127:112]
    g1d[4] = 32;                     // tile_dim1=32           @ bits[143:128]
    g1d[5] = 128;                    // tensor_dim0_stride=128 @ bits[191:160]
    g1d[6] = 0; g1d[7] = 0;
    v4i_ gz = {0, 0, 0, 0};
#if __clang_major__ >= 23
    v8i_ gz8 = {0, 0, 0, 0, 0, 0, 0, 0};
    __builtin_amdgcn_tensor_load_to_lds(g0, g1d, gz, gz, gz8, 0);
#else
    __builtin_amdgcn_tensor_load_to_lds(g0, g1d, gz, gz, 0);
#endif
    __builtin_amdgcn_s_wait_tensorcnt(0);
  }
#else
  {
    const uint4* src = reinterpret_cast<const uint4*>(Xb + (size_t)r0 * F_DIM);
    uint4* dst = reinterpret_cast<uint4*>(&Xs[0][0]);
    for (int i = lane; i < (32 * F_DIM) / 8; i += 32) dst[i] = src[i];
  }
#endif
  __syncthreads();

  // ---- GEMM1: H = X @ W1^T, fused bias + BN(eval) + LeakyReLU(0.1) ----
  // all 8 A fragments (4 K-steps x 2 M-strips) live in registers
  v16bf a1[2][4];
  #pragma unroll
  for (int m = 0; m < 2; ++m)
    #pragma unroll
    for (int kk = 0; kk < 4; ++kk)
      a1[m][kk] = load_frag_a(&Xs[m * 16 + lm][0], hi, kk * 32);

  for (int ct = 0; ct < H_DIM / 16; ++ct) {
    int j = ct * 16 + lm;                     // hidden unit owned by this lane
    const __bf16* brow = W1b + (size_t)j * F_DIM;
    v8f acc0 = {}, acc1 = {};
    #pragma unroll
    for (int kk = 0; kk < F_DIM / 32; ++kk) {
      v16bf b = load_frag_b(brow, hi, kk * 32);    // loaded once, used twice
      acc0 = wmma_bf16(a1[0][kk], b, acc0);
      acc1 = wmma_bf16(a1[1][kk], b, acc1);
    }
    // fold bias + BN into one FMA: h = c*mul + add, then leaky relu
    float mul = rsqrtf(rv[j] + BN_EPS) * g1[j];
    float add = (b1[j] - rm[j]) * mul + be1[j];
    #pragma unroll
    for (int r = 0; r < 8; ++r) {
      float h0 = acc0[r] * mul + add;
      h0 = h0 >= 0.0f ? h0 : 0.1f * h0;
      Hs[r + hi * 8][j] = (__bf16)h0;         // C layout: M = r + 8*hi, N = lane%16
      float h1 = acc1[r] * mul + add;
      h1 = h1 >= 0.0f ? h1 : 0.1f * h1;
      Hs[16 + r + hi * 8][j] = (__bf16)h1;
    }
  }
  __syncthreads();

  // ---- GEMM2: E = H @ W2^T + b2 (hold all 16 accumulators, K-step outer) ----
  v8f acc[F_DIM / 16][2];
  {
    v8f z = {};
    #pragma unroll
    for (int ct = 0; ct < F_DIM / 16; ++ct) { acc[ct][0] = z; acc[ct][1] = z; }
  }
  __builtin_prefetch(W2b, 0, 1);
  for (int kk = 0; kk < H_DIM / 32; ++kk) {
    v16bf a0 = load_frag_a(&Hs[lm][0], hi, kk * 32);
    v16bf a1v = load_frag_a(&Hs[16 + lm][0], hi, kk * 32);
    #pragma unroll
    for (int ct = 0; ct < F_DIM / 16; ++ct) {
      const __bf16* brow = W2b + (size_t)(ct * 16 + lm) * H_DIM;
      v16bf b = load_frag_b(brow, hi, kk * 32);    // loaded once, used twice
      acc[ct][0] = wmma_bf16(a0, b, acc[ct][0]);
      acc[ct][1] = wmma_bf16(a1v, b, acc[ct][1]);
    }
  }
  #pragma unroll
  for (int ct = 0; ct < F_DIM / 16; ++ct) {
    int f = ct * 16 + lm;
    float bf = b2[f];
    #pragma unroll
    for (int r = 0; r < 8; ++r) {
      E[(size_t)(r0 + r + hi * 8) * F_DIM + f]      = acc[ct][0][r] + bf;
      E[(size_t)(r0 + 16 + r + hi * 8) * F_DIM + f] = acc[ct][1][r] + bf;
    }
  }
}

// --------------------------- kernel 4: distances + sumsq -------------------
__global__ __launch_bounds__(256) void dist_kernel(const float* __restrict__ E,
                                                   float* __restrict__ lt,
                                                   float* __restrict__ pss, int N) {
  __shared__ float red[256];
  int n = blockIdx.x * 256 + threadIdx.x;
  const float* eu = E + (size_t)n * F_DIM;
  const float* ep = E + (size_t)(N + n) * F_DIM;
  const float* en = E + (size_t)(2 * N + n) * F_DIM;
  float dp = 0.f, dn = 0.f, ss = 0.f;
  for (int f = 0; f < F_DIM; ++f) {
    float a = eu[f], b = ep[f], c = en[f];
    float x = a - b, y = a - c;
    dp += x * x; dn += y * y;
    ss += a * a + b * b + c * c;
  }
  float l = dp - dn + MARGIN;
  lt[n] = l > 0.f ? l : 0.f;
  red[threadIdx.x] = ss;
  __syncthreads();
  for (int s = 128; s > 0; s >>= 1) {
    if (threadIdx.x < s) red[threadIdx.x] += red[threadIdx.x + s];
    __syncthreads();
  }
  if (threadIdx.x == 0) pss[blockIdx.x] = red[0];
}

// --------------------------- kernel 5: group reweighting -------------------
// M[n] = sum_m ind[m] * [ (user,pos)[m] == (user,pos)[n] ]  (== segment_sum)
__global__ __launch_bounds__(256) void group_kernel(const int* __restrict__ u,
                                                    const int* __restrict__ p,
                                                    const float* __restrict__ lt,
                                                    const int* __restrict__ Jp,
                                                    const int* __restrict__ Up,
                                                    float* __restrict__ pwl, int N) {
  __shared__ int   su[256];
  __shared__ int   sp[256];
  __shared__ float sa[256];
  __shared__ float red[256];
  int t = threadIdx.x;
  int n = blockIdx.x * 256 + t;
  int un = u[n], pn = p[n];
  float ltn = lt[n];
  float cnt = 0.f;
  int chunks = N / 256;
  for (int c = 0; c < chunks; ++c) {
    int i = c * 256 + t;
    su[t] = u[i]; sp[t] = p[i];
    sa[t] = lt[i] > 0.f ? 1.f : 0.f;
    __syncthreads();
    #pragma unroll 8
    for (int k = 0; k < 256; ++k)
      if (su[k] == un && sp[k] == pn) cnt += sa[k];
    __syncthreads();
  }
  float Jf = (float)Jp[0], Uf = (float)Up[0];
  float rank = Uf > 0.f ? Jf * cnt / Uf : 0.f;
  float w = logf(rank + 1.f);                  // stop_gradient constant
  red[t] = ltn * w;
  __syncthreads();
  for (int s = 128; s > 0; s >>= 1) {
    if (t < s) red[t] += red[t + s];
    __syncthreads();
  }
  if (t == 0) pwl[blockIdx.x] = red[0];
}

// --------------------------- kernel 6: finalize ----------------------------
__global__ void finalize_kernel(const float* __restrict__ pss,
                                const float* __restrict__ pwl,
                                float* __restrict__ out, int nb, int N) {
  float ss = 0.f, wl = 0.f;
  for (int i = 0; i < nb; ++i) { ss += pss[i]; wl += pwl[i]; }
  float mean = wl / (float)N;
  float l2   = L2_REG * ss / (float)N;
  out[0] = mean + l2;      // loss
  out[1] = mean;           // loss_triplet_mean
}

// ---------------------------------------------------------------------------
extern "C" void kernel_launch(void* const* d_in, const int* in_sizes, int n_in,
                              void* d_out, int out_size, void* d_ws, size_t ws_size,
                              hipStream_t stream) {
  const int*   user0 = (const int*)d_in[0];
  const int*   posi  = (const int*)d_in[1];
  const int*   negi  = (const int*)d_in[2];
  /* d_in[3] = ratings (unused by reference) */
  const float* utab  = (const float*)d_in[4];
  const float* itab  = (const float*)d_in[5];
  const float* W1    = (const float*)d_in[6];
  const float* b1    = (const float*)d_in[7];
  const float* g1    = (const float*)d_in[8];
  const float* be1   = (const float*)d_in[9];
  const float* rm    = (const float*)d_in[10];
  const float* rv    = (const float*)d_in[11];
  const float* W2    = (const float*)d_in[12];
  const float* b2    = (const float*)d_in[13];
  const int*   Jp    = (const int*)d_in[14];
  const int*   Up    = (const int*)d_in[15];
  (void)n_in; (void)out_size; (void)ws_size;

  const int N  = in_sizes[0];        // 8192
  const int nb = N / 256;            // partial count (32)

  char* ws = (char*)d_ws;
  size_t off = 0;
  auto alloc = [&](size_t bytes) -> void* {
    void* q = ws + off;
    off = (off + bytes + 255) & ~(size_t)255;
    return q;
  };
  __bf16* Xb  = (__bf16*)alloc((size_t)3 * N * F_DIM * sizeof(__bf16));
  __bf16* W1b = (__bf16*)alloc((size_t)H_DIM * F_DIM * sizeof(__bf16));
  __bf16* W2b = (__bf16*)alloc((size_t)F_DIM * H_DIM * sizeof(__bf16));
  float*  E   = (float*)alloc((size_t)3 * N * F_DIM * sizeof(float));
  float*  lt  = (float*)alloc((size_t)N * sizeof(float));
  float*  pss = (float*)alloc((size_t)nb * sizeof(float));
  float*  pwl = (float*)alloc((size_t)nb * sizeof(float));

  wconv_kernel<<<(H_DIM * F_DIM + 255) / 256, 256, 0, stream>>>(W1, W2, W1b, W2b);
  gather_kernel<<<(3 * N) / 8, 256, 0, stream>>>(user0, posi, negi, utab, itab, Xb, N);
  filter_kernel<<<(3 * N) / 32, 32, 0, stream>>>(Xb, W1b, W2b, b1, g1, be1, rm, rv, b2, E);
  dist_kernel<<<nb, 256, 0, stream>>>(E, lt, pss, N);
  group_kernel<<<nb, 256, 0, stream>>>(user0, posi, lt, Jp, Up, pwl, N);
  finalize_kernel<<<1, 1, 0, stream>>>(pss, pwl, (float*)d_out, nb, N);
}
